// TaskAwareMoEActor_33157147525373
// MI455X (gfx1250) — matrix operand order repaired and checked by reading, not compiled, not run on hardware
//
#include <hip/hip_runtime.h>
#include <hip/hip_bf16.h>

typedef _Float16 half8 __attribute__((ext_vector_type(8)));
typedef _Float16 v16h  __attribute__((ext_vector_type(16)));
typedef float    v8f   __attribute__((ext_vector_type(8)));

// Tensor Data Mover descriptor vectors (6-arg builtin: clang-23/therock-10.0)
#define USE_TDM 1
typedef unsigned int u32x4 __attribute__((ext_vector_type(4)));
typedef int          i32x4 __attribute__((ext_vector_type(4)));
typedef int          i32x8 __attribute__((ext_vector_type(8)));

__device__ __forceinline__ v16h h16_combine(half8 lo, half8 hi) {
    v16h r;
#pragma unroll
    for (int i = 0; i < 8; ++i) { r[i] = lo[i]; r[i + 8] = hi[i]; }
    return r;
}

// ---------------------------------------------------------------------------
// Fused Linear(+bias) + ELU (+LayerNorm) via v_wmma_f32_16x16x32_f16.
//
// Block = NW wave32s. Each block owns one (expert, MT*16-row panel) and ALL
// N columns (so LN fuses in-block). Wave w owns column tiles [w*NT, w*NT+NT).
// Per kc chunk: B-fragments loaded once per wave and reused across MT row
// tiles. A panel DMA'd into LDS by the Tensor Data Mover (TENSORcnt), then
// read as WMMA A-fragments by all waves.
//
// Fragment layouts (CDNA5 ISA 7.12.2):
//  A 16x32 f16 : lanes 0-15: M=lane, K={0..7} in v0-3, {16..23} in v4-7;
//                lanes 16-31: M=lane-16, K offsets +8.
//  B 32x16 f16 : lanes 0-15: N=lane, K={0..7,16..23}; lanes 16-31: K offsets +8.
//  C/D 16x16   : vgpr r, lanes 0-15: M=r, N=lane; lanes 16-31: M=r+8.
// ---------------------------------------------------------------------------
template <int K_DIM, int N_DIM, int MT, int NW, bool DO_LN>
__global__ __launch_bounds__(NW * 32)
void moe_layer_wmma(const _Float16* __restrict__ A, int lda, unsigned long long a_estride,
                    const _Float16* __restrict__ Wt,   // [E][N][K] f16
                    const float* __restrict__ bias,    // [E][N]
                    const float* __restrict__ gamma,   // [E][N] (LN only)
                    const float* __restrict__ beta,    // [E][N] (LN only)
                    _Float16* __restrict__ Out, int ld_out, int col_off,
                    unsigned long long out_estride, int Bsz) {
    static_assert(K_DIM % 32 == 0, "K multiple of 32");
    constexpr int NT   = N_DIM / (NW * 16);
    constexpr int ROWS = MT * 16;
    static_assert(NT >= 1 && NT * NW * 16 == N_DIM, "column tiling");

    const int e        = blockIdx.y;
    const int rowBlock = blockIdx.x;

    const _Float16* Ab = A + (unsigned long long)e * a_estride + (size_t)rowBlock * ROWS * lda;
    const _Float16* We = Wt + (size_t)e * N_DIM * K_DIM;
    const float* be = bias  + (size_t)e * N_DIM;
    const float* ge = gamma ? gamma + (size_t)e * N_DIM : gamma;
    const float* he = beta  ? beta  + (size_t)e * N_DIM : beta;

    __shared__ _Float16 sA[ROWS * K_DIM];
    __shared__ float sSum[ROWS], sSq[ROWS];

    if ((int)threadIdx.x < ROWS) { sSum[threadIdx.x] = 0.f; sSq[threadIdx.x] = 0.f; }

    const int lane = threadIdx.x & 31;
    const int wave = threadIdx.x >> 5;

#if USE_TDM
    // ---- Tensor Data Mover: DMA the ROWS x K_DIM (2-byte) tile into LDS ----
    // D# per CDNA5 ISA: group0 {count, lds_addr,
    // global_addr, type=2}; group1 {data_size=2B, tensor_dim0=K,
    // tensor_dim1=ROWS, tile_dim0=K, tile_dim1=ROWS, dim0_stride=lda}.
    if (wave == 0) {                 // single wave issues; EXEC ignored by TDM
        unsigned long long ga = (unsigned long long)(uintptr_t)Ab;
        unsigned int ldsa = (unsigned int)(uintptr_t)&sA[0];
        u32x4 g0;
        g0[0] = 1u;                                              // count=1 (valid), user mode
        g0[1] = ldsa;                                            // lds_addr (bytes)
        g0[2] = (unsigned int)(ga & 0xFFFFFFFFu);                // global_addr[31:0]
        g0[3] = (unsigned int)((ga >> 32) & 0x1FFFFFFu) | (2u << 30); // addr[56:32] | type=2
        i32x8 g1;
        g1[0] = 0x00010000;                                      // data_size=1 (2 bytes)
        g1[1] = (int)((K_DIM & 0xFFFF) << 16);                   // tensor_dim0[15:0] @ bits 63:48
        g1[2] = (int)(((unsigned)K_DIM >> 16) | ((ROWS & 0xFFFF) << 16)); // td0 hi | td1 lo
        g1[3] = (int)(((unsigned)ROWS >> 16) | ((unsigned)K_DIM << 16));  // td1 hi | tile_dim0=K
        g1[4] = (int)(ROWS & 0xFFFF);                            // tile_dim1=ROWS, tile_dim2=0
        g1[5] = (int)lda;                                        // tensor_dim0_stride[31:0]
        g1[6] = 0;
        g1[7] = 0;
        i32x4 gz4 = {0, 0, 0, 0};                                // groups 2/3: unused (2D tile)
        i32x8 gz8 = {0, 0, 0, 0, 0, 0, 0, 0};
        __builtin_amdgcn_tensor_load_to_lds(g0, g1, gz4, gz4, gz8, 0);
        __builtin_amdgcn_s_wait_tensorcnt(0);
    }
    __syncthreads();
#else
    // cooperative stage of the ROWS x K activation panel (b128 chunks)
    constexpr int CPR = K_DIM / 8;
    for (int c = threadIdx.x; c < ROWS * CPR; c += NW * 32) {
        int r = c / CPR, k8 = c % CPR;
        half8 v = *(const half8*)(Ab + (size_t)r * lda + k8 * 8);
        *(half8*)(&sA[r * K_DIM + k8 * 8]) = v;
    }
    __syncthreads();
#endif

    const int nloc = lane & 15;
    const int hi   = lane >> 4;
    const int khA  = hi * 8;
    const int khB  = hi * 16;

    v8f acc[MT][NT];
#pragma unroll
    for (int mt = 0; mt < MT; ++mt)
#pragma unroll
        for (int t = 0; t < NT; ++t)
#pragma unroll
            for (int r = 0; r < 8; ++r) acc[mt][t][r] = 0.f;

    for (int kc = 0; kc < K_DIM / 32; ++kc) {
        // load B fragments once per kc; reuse across MT row tiles
        v16h bfr[NT];
#pragma unroll
        for (int t = 0; t < NT; ++t) {
            int col = (wave * NT + t) * 16 + nloc;
            const _Float16* bp = We + (size_t)col * K_DIM + kc * 32 + khA;
            bfr[t] = h16_combine(*(const half8*)bp, *(const half8*)(bp + 16));
            __builtin_prefetch((const void*)(bp + 32), 0, 3);  // next kc chunk -> global_prefetch_b8
        }
#pragma unroll
        for (int mt = 0; mt < MT; ++mt) {
            const _Float16* ap = &sA[(mt * 16 + nloc) * K_DIM + kc * 32 + khA];
            v16h a = h16_combine(*(const half8*)ap, *(const half8*)(ap + 16));
#pragma unroll
            for (int t = 0; t < NT; ++t)
                acc[mt][t] = __builtin_amdgcn_wmma_f32_16x16x32_f16(
                    false, a, false, bfr[t], (short)0, acc[mt][t], false, false);
        }
    }

    // ---- epilogue: +bias, ELU in place; optional LN over full row ----
#pragma unroll
    for (int mt = 0; mt < MT; ++mt)
#pragma unroll
        for (int t = 0; t < NT; ++t) {
            int col = (wave * NT + t) * 16 + nloc;
            float bia = be[col];
#pragma unroll
            for (int r = 0; r < 8; ++r) {
                float v = acc[mt][t][r] + bia;
                acc[mt][t][r] = (v > 0.f) ? v : (__expf(v) - 1.f);   // ELU
            }
        }

    if (DO_LN) {
#pragma unroll
        for (int mt = 0; mt < MT; ++mt) {
            float s[8], q[8];
#pragma unroll
            for (int r = 0; r < 8; ++r) { s[r] = 0.f; q[r] = 0.f; }
#pragma unroll
            for (int t = 0; t < NT; ++t)
#pragma unroll
                for (int r = 0; r < 8; ++r) {
                    float v = acc[mt][t][r];
                    s[r] += v; q[r] += v * v;
                }
#pragma unroll
            for (int m = 1; m < 16; m <<= 1)
#pragma unroll
                for (int r = 0; r < 8; ++r) {
                    s[r] += __shfl_xor(s[r], m, 32);
                    q[r] += __shfl_xor(q[r], m, 32);
                }
            if (nloc == 0) {
#pragma unroll
                for (int r = 0; r < 8; ++r) {
                    atomicAdd(&sSum[mt * 16 + r + 8 * hi], s[r]);   // ds_add_f32
                    atomicAdd(&sSq [mt * 16 + r + 8 * hi], q[r]);
                }
            }
        }
        __syncthreads();
    }

    const unsigned long long outBase = (unsigned long long)e * out_estride;
#pragma unroll
    for (int mt = 0; mt < MT; ++mt) {
#pragma unroll
        for (int t = 0; t < NT; ++t) {
            int col = (wave * NT + t) * 16 + nloc;
            float g = 1.f, h = 0.f;
            if (DO_LN) { g = ge[col]; h = he[col]; }
#pragma unroll
            for (int r = 0; r < 8; ++r) {
                int M = mt * 16 + r + 8 * hi;
                int brow = rowBlock * ROWS + M;
                float v = acc[mt][t][r];
                if (DO_LN) {
                    float mu  = sSum[M] * (1.0f / N_DIM);
                    float var = sSq[M]  * (1.0f / N_DIM) - mu * mu;
                    v = (v - mu) * rsqrtf(var + 1e-5f) * g + h;
                }
                Out[outBase + (size_t)brow * ld_out + col_off + col] = (_Float16)v;
            }
        }
    }
}

// ---------------------------------------------------------------------------
// Helpers
// ---------------------------------------------------------------------------

// f32 [EE][K][N] -> f16 transposed [EE][N][K]
__global__ void wconv_kernel(const float* __restrict__ src, _Float16* __restrict__ dst,
                             int K, int N, long total) {
    long i = (long)blockIdx.x * blockDim.x + threadIdx.x;
    if (i >= total) return;
    int n = (int)(i % N);
    long r = i / N;
    int k = (int)(r % K);
    long e = r / K;
    dst[(e * N + n) * (long)K + k] = (_Float16)src[i];
}

// Build concat [B][608] f16 (state 0..95, vlm 96..607)
__global__ void build_g0_kernel(const float* __restrict__ st, const float* __restrict__ vl,
                                _Float16* __restrict__ G0, int Bsz) {
    long i = (long)blockIdx.x * blockDim.x + threadIdx.x;
    long total = (long)Bsz * 608;
    if (i >= total) return;
    long b = i / 608;
    int  c = (int)(i % 608);
    float v = (c < 96) ? st[b * 96 + c] : vl[b * 512 + (c - 96)];
    G0[i] = (_Float16)v;
}

__global__ void zero16_kernel(float* p) {
    if (blockIdx.x == 0 && threadIdx.x < 16) p[threadIdx.x] = 0.f;
}

// final expert head: [E][B][256] f16 x fw3[E][256][12] + fb3 -> [E][B][12] f32
__global__ void fuse3_kernel(const _Float16* __restrict__ H, const float* __restrict__ fw3,
                             const float* __restrict__ fb3, float* __restrict__ OutE, int Bsz) {
    long t = (long)blockIdx.x * blockDim.x + threadIdx.x;
    long pair = t >> 4;
    int  d = (int)(t & 15);
    long rows = (long)Bsz * 8;
    if (pair >= rows || d >= 12) return;
    int e = (int)(pair / Bsz);
    const _Float16* hr = H + pair * 256;
    const float* wp = fw3 + (size_t)e * 256 * 12 + d;
    float acc = fb3[e * 12 + d];
    for (int k = 0; k < 256; ++k) acc += (float)hr[k] * wp[(size_t)k * 12];
    OutE[pair * 12 + d] = acc;
}

// gate head: g2[B][128] x gw3[128][8] + gb3, softmax, top-2, aux accumulation
__global__ void gate_final_kernel(const _Float16* __restrict__ g2, const float* __restrict__ gw3,
                                  const float* __restrict__ gb3,
                                  float* __restrict__ accP, float* __restrict__ accC,
                                  int* __restrict__ tIdx, float* __restrict__ tW,
                                  float* __restrict__ outIdxF, float* __restrict__ outScore,
                                  int Bsz) {
    __shared__ float lp[8], lc[8];
    if (threadIdx.x < 8) { lp[threadIdx.x] = 0.f; lc[threadIdx.x] = 0.f; }
    __syncthreads();
    int b = blockIdx.x * blockDim.x + threadIdx.x;
    if (b < Bsz) {
        float lo[8];
#pragma unroll
        for (int j = 0; j < 8; ++j) lo[j] = gb3[j];
        const _Float16* gr = g2 + (size_t)b * 128;
        for (int k = 0; k < 128; ++k) {
            float x = (float)gr[k];
#pragma unroll
            for (int j = 0; j < 8; ++j) lo[j] += x * gw3[k * 8 + j];
        }
        float mx = lo[0];
#pragma unroll
        for (int j = 1; j < 8; ++j) mx = fmaxf(mx, lo[j]);
        float p[8]; float ssum = 0.f;
#pragma unroll
        for (int j = 0; j < 8; ++j) { p[j] = __expf(lo[j] - mx); ssum += p[j]; }
        float inv = 1.f / ssum;
#pragma unroll
        for (int j = 0; j < 8; ++j) p[j] *= inv;
        int i0 = 0; float v0 = p[0];
#pragma unroll
        for (int j = 1; j < 8; ++j) if (p[j] > v0) { v0 = p[j]; i0 = j; }
        int i1 = -1; float v1 = -1.f;
#pragma unroll
        for (int j = 0; j < 8; ++j) if (j != i0 && p[j] > v1) { v1 = p[j]; i1 = j; }
        float wsum = v0 + v1 + 1e-9f;
        tIdx[b * 2] = i0; tIdx[b * 2 + 1] = i1;
        tW[b * 2] = v0 / wsum; tW[b * 2 + 1] = v1 / wsum;
        outIdxF[b * 2] = (float)i0; outIdxF[b * 2 + 1] = (float)i1;
        outScore[b * 2] = v0; outScore[b * 2 + 1] = v1;
#pragma unroll
        for (int j = 0; j < 8; ++j) atomicAdd(&lp[j], p[j]);
        atomicAdd(&lc[i0], 1.f);
        atomicAdd(&lc[i1], 1.f);
    }
    __syncthreads();
    if (threadIdx.x < 8) {
        atomicAdd(&accP[threadIdx.x], lp[threadIdx.x]);
        atomicAdd(&accC[threadIdx.x], lc[threadIdx.x]);
    }
}

// weighted top-2 mix -> action [B][12]
__global__ void combine_kernel(const float* __restrict__ OutE, const int* __restrict__ tIdx,
                               const float* __restrict__ tW, float* __restrict__ act, int Bsz) {
    long t = (long)blockIdx.x * blockDim.x + threadIdx.x;
    long b = t >> 4;
    int  d = (int)(t & 15);
    if (b >= Bsz || d >= 12) return;
    int i0 = tIdx[b * 2], i1 = tIdx[b * 2 + 1];
    float w0 = tW[b * 2], w1 = tW[b * 2 + 1];
    act[b * 12 + d] = w0 * OutE[((size_t)i0 * Bsz + b) * 12 + d]
                    + w1 * OutE[((size_t)i1 * Bsz + b) * 12 + d];
}

__global__ void finalize_aux_kernel(const float* __restrict__ accP, const float* __restrict__ accC,
                                    float* __restrict__ out_aux, int Bsz) {
    if (blockIdx.x == 0 && threadIdx.x == 0) {
        float aux = 0.f;
        for (int e = 0; e < 8; ++e) {
            float f = accC[e] / ((float)Bsz * 2.f);
            float P = accP[e] / (float)Bsz;
            aux += f * P;
        }
        *out_aux = 8.f * aux;
    }
}

// ---------------------------------------------------------------------------
extern "C" void kernel_launch(void* const* d_in, const int* in_sizes, int n_in,
                              void* d_out, int out_size, void* d_ws, size_t ws_size,
                              hipStream_t stream) {
    const float* state = (const float*)d_in[0];
    const float* vlm   = (const float*)d_in[1];
    const float* sw1 = (const float*)d_in[2];  const float* sb1 = (const float*)d_in[3];
    const float* sg1 = (const float*)d_in[4];  const float* sh1 = (const float*)d_in[5];
    const float* sw2 = (const float*)d_in[6];  const float* sb2 = (const float*)d_in[7];
    const float* sg2 = (const float*)d_in[8];  const float* sh2 = (const float*)d_in[9];
    const float* vw1 = (const float*)d_in[10]; const float* vb1 = (const float*)d_in[11];
    const float* vg1 = (const float*)d_in[12]; const float* vh1 = (const float*)d_in[13];
    const float* vw2 = (const float*)d_in[14]; const float* vb2 = (const float*)d_in[15];
    const float* vg2 = (const float*)d_in[16]; const float* vh2 = (const float*)d_in[17];
    const float* fw1 = (const float*)d_in[18]; const float* fb1 = (const float*)d_in[19];
    const float* fg1 = (const float*)d_in[20]; const float* fh1 = (const float*)d_in[21];
    const float* fw2 = (const float*)d_in[22]; const float* fb2 = (const float*)d_in[23];
    const float* fg2 = (const float*)d_in[24]; const float* fh2 = (const float*)d_in[25];
    const float* fw3 = (const float*)d_in[26]; const float* fb3 = (const float*)d_in[27];
    const float* gw1 = (const float*)d_in[28]; const float* gb1 = (const float*)d_in[29];
    const float* gw2 = (const float*)d_in[30]; const float* gb2 = (const float*)d_in[31];
    const float* gw3 = (const float*)d_in[32]; const float* gb3 = (const float*)d_in[33];

    const int Bsz = in_sizes[0] / 96;     // 32768
    const int E = 8;

    // ---- workspace carve-out (256B aligned) ----
    char* wsb = (char*)d_ws;
    size_t off = 0;
    auto alloc = [&](size_t bytes) -> void* {
        void* p = wsb + off;
        off = (off + bytes + 255) & ~(size_t)255;
        return p;
    };
    _Float16* G0   = (_Float16*)alloc((size_t)Bsz * 608 * 2);      // concat input f16
    _Float16* Hbuf = (_Float16*)alloc((size_t)E * Bsz * 256 * 2);  // s_h1 -> f_h2
    _Float16* A512 = (_Float16*)alloc((size_t)E * Bsz * 512 * 2);  // v_h1 -> f_h1
    _Float16* Fbuf = (_Float16*)alloc((size_t)E * Bsz * 384 * 2);  // fused
    float*    OutE = (float*)   alloc((size_t)E * Bsz * 12 * 4);   // expert outputs
    _Float16* g1   = (_Float16*)alloc((size_t)Bsz * 256 * 2);
    _Float16* g2   = (_Float16*)alloc((size_t)Bsz * 128 * 2);
    _Float16* sw1T = (_Float16*)alloc((size_t)E * 96  * 256 * 2);
    _Float16* sw2T = (_Float16*)alloc((size_t)E * 256 * 128 * 2);
    _Float16* vw1T = (_Float16*)alloc((size_t)E * 512 * 512 * 2);
    _Float16* vw2T = (_Float16*)alloc((size_t)E * 512 * 256 * 2);
    _Float16* fw1T = (_Float16*)alloc((size_t)E * 384 * 512 * 2);
    _Float16* fw2T = (_Float16*)alloc((size_t)E * 512 * 256 * 2);
    _Float16* gw1T = (_Float16*)alloc((size_t)608 * 256 * 2);
    _Float16* gw2T = (_Float16*)alloc((size_t)256 * 128 * 2);
    float*    accs = (float*)alloc(16 * 4);                        // accP[8] + accC[8]
    int*      tIdx = (int*)  alloc((size_t)Bsz * 2 * 4);
    float*    tW   = (float*)alloc((size_t)Bsz * 2 * 4);
    float* accP = accs, *accC = accs + 8;

    // ---- d_out layout: action[B*12], aux[1], idx[B*2] (as float), scores[B*2]
    float* out_action = (float*)d_out;
    float* out_aux    = out_action + (size_t)Bsz * 12;
    float* out_idx    = out_aux + 1;
    float* out_score  = out_idx + (size_t)Bsz * 2;

    const dim3 blk(256);

    // zero per-call accumulators
    zero16_kernel<<<1, 32, 0, stream>>>(accs);

    // weight f32 -> f16 transposed
    auto wc = [&](const float* src, _Float16* dst, int K, int N, int EE) {
        long total = (long)EE * K * N;
        wconv_kernel<<<(unsigned)((total + 255) / 256), blk, 0, stream>>>(src, dst, K, N, total);
    };
    wc(sw1, sw1T, 96, 256, E);  wc(sw2, sw2T, 256, 128, E);
    wc(vw1, vw1T, 512, 512, E); wc(vw2, vw2T, 512, 256, E);
    wc(fw1, fw1T, 384, 512, E); wc(fw2, fw2T, 512, 256, E);
    wc(gw1, gw1T, 608, 256, 1); wc(gw2, gw2T, 256, 128, 1);

    { // concat input f16
        long total = (long)Bsz * 608;
        build_g0_kernel<<<(unsigned)((total + 255) / 256), blk, 0, stream>>>(state, vlm, G0, Bsz);
    }

    // 64 batch rows per block (MT=4)
    const dim3 grdE(Bsz / 64, E);
    const dim3 grd1(Bsz / 64, 1);
    const dim3 blk512(512), blk256(256);
    typedef unsigned long long ull;

    // gate MLP hidden layers (ELU, no LN)
    moe_layer_wmma<608, 256, 4, 16, false><<<grd1, blk512, 0, stream>>>(
        G0, 608, 0ull, gw1T, gb1, nullptr, nullptr, g1, 256, 0, 0ull, Bsz);
    moe_layer_wmma<256, 128, 4, 8, false><<<grd1, blk256, 0, stream>>>(
        g1, 256, 0ull, gw2T, gb2, nullptr, nullptr, g2, 128, 0, 0ull, Bsz);
    gate_final_kernel<<<(Bsz + 255) / 256, blk, 0, stream>>>(
        g2, gw3, gb3, accP, accC, tIdx, tW, out_idx, out_score, Bsz);

    // state encoder: 96 -> 256 -> 128 (into fused cols 0..127)
    moe_layer_wmma<96, 256, 4, 16, true><<<grdE, blk512, 0, stream>>>(
        G0, 608, 0ull, sw1T, sb1, sg1, sh1, Hbuf, 256, 0, (ull)Bsz * 256, Bsz);
    moe_layer_wmma<256, 128, 4, 8, true><<<grdE, blk256, 0, stream>>>(
        Hbuf, 256, (ull)Bsz * 256, sw2T, sb2, sg2, sh2, Fbuf, 384, 0, (ull)Bsz * 384, Bsz);

    // vlm encoder: 512 -> 512 -> 256 (into fused cols 128..383)
    moe_layer_wmma<512, 512, 4, 16, true><<<grdE, blk512, 0, stream>>>(
        G0 + 96, 608, 0ull, vw1T, vb1, vg1, vh1, A512, 512, 0, (ull)Bsz * 512, Bsz);
    moe_layer_wmma<512, 256, 4, 16, true><<<grdE, blk512, 0, stream>>>(
        A512, 512, (ull)Bsz * 512, vw2T, vb2, vg2, vh2, Fbuf, 384, 128, (ull)Bsz * 384, Bsz);

    // fusion trunk: 384 -> 512 -> 256
    moe_layer_wmma<384, 512, 4, 16, true><<<grdE, blk512, 0, stream>>>(
        Fbuf, 384, (ull)Bsz * 384, fw1T, fb1, fg1, fh1, A512, 512, 0, (ull)Bsz * 512, Bsz);
    moe_layer_wmma<512, 256, 4, 16, true><<<grdE, blk512, 0, stream>>>(
        A512, 512, (ull)Bsz * 512, fw2T, fb2, fg2, fh2, Hbuf, 256, 0, (ull)Bsz * 256, Bsz);

    // expert head 256 -> 12 (f32)
    { long total = (long)E * Bsz * 16;
      fuse3_kernel<<<(unsigned)((total + 255) / 256), blk, 0, stream>>>(Hbuf, fw3, fb3, OutE, Bsz); }

    // top-2 weighted combine + aux loss
    { long total = (long)Bsz * 16;
      combine_kernel<<<(unsigned)((total + 255) / 256), blk, 0, stream>>>(OutE, tIdx, tW, out_action, Bsz); }
    finalize_aux_kernel<<<1, 32, 0, stream>>>(accP, accC, out_aux, Bsz);
}
